// SAGE_80977313399688
// MI455X (gfx1250) — matrix operand order repaired and checked by reading, not compiled, not run on hardware
//
#include <hip/hip_runtime.h>
#include <hip/hip_bf16.h>

typedef unsigned short u16b;
typedef unsigned int   u32b;
typedef __attribute__((ext_vector_type(16))) __bf16 v16bf;
typedef __attribute__((ext_vector_type(8)))  float  v8f;

union Frag { u32b u[8]; v16bf v; };

__device__ __forceinline__ float bflo(u32b u) { return __uint_as_float(u << 16); }
__device__ __forceinline__ float bfhi(u32b u) { return __uint_as_float(u & 0xffff0000u); }
__device__ __forceinline__ u16b  f2bf(float f) {
    u32b u = __float_as_uint(f);
    u32b r = (u + 0x7fffu + ((u >> 16) & 1u)) >> 16;   // round-to-nearest-even
    return (u16b)r;
}

// ---------------------------------------------------------------- utilities
__global__ void zero_f32(float* __restrict__ p, long long n) {
    long long i = (long long)blockIdx.x * blockDim.x + threadIdx.x;
    long long stride = (long long)gridDim.x * blockDim.x;
    for (; i < n; i += stride) p[i] = 0.0f;
}

__global__ void deg_count(const int* __restrict__ dst, float* __restrict__ deg, int E) {
    int i = blockIdx.x * blockDim.x + threadIdx.x;
    if (i < E) atomicAdd(&deg[dst[i]], 1.0f);
}

__global__ void deg_inv(float* __restrict__ deg, int n) {
    int i = blockIdx.x * blockDim.x + threadIdx.x;
    if (i < n) deg[i] = 1.0f / fmaxf(deg[i], 1.0f);
}

// transpose+convert weight: W[k][Fout] fp32 -> WT[n][k] bf16, zero-padded to FoutPad rows
__global__ void wprep(const float* __restrict__ W, u16b* __restrict__ WT,
                      int K, int Fout, int FoutPad) {
    int idx = blockIdx.x * blockDim.x + threadIdx.x;
    int tot = K * FoutPad;
    if (idx >= tot) return;
    int n = idx / K, k = idx - n * K;
    WT[idx] = (n < Fout) ? f2bf(W[(size_t)k * Fout + n]) : (u16b)0;
}

__global__ void cvt_bf16(const float* __restrict__ x, u16b* __restrict__ xB, long long total) {
    long long i = (long long)blockIdx.x * blockDim.x + threadIdx.x;
    long long stride = (long long)gridDim.x * blockDim.x;
    for (; i < total; i += stride) xB[i] = f2bf(x[i]);
}

// agg[i] *= inv_deg[row]; convert to bf16
__global__ void scale_cvt(const float* __restrict__ agg, const float* __restrict__ inv,
                          u16b* __restrict__ aggB, long long total) {
    long long i = (long long)blockIdx.x * blockDim.x + threadIdx.x;
    long long stride = (long long)gridDim.x * blockDim.x;
    for (; i < total; i += stride) {
        long long row = i >> 7;
        aggB[i] = f2bf(agg[i] * inv[row]);
    }
}

// ------------------------------------------------------------- scatter-add
// one wave per edge: read h[src] row (128 bf16 = 64 dwords, coalesced),
// fp32 atomic-accumulate into agg[dst] row.
__global__ __launch_bounds__(256) void scatter_add(
    const int* __restrict__ src, const int* __restrict__ dst,
    const u16b* __restrict__ hB, float* __restrict__ agg, int E) {
    int e = blockIdx.x * 8 + (threadIdx.x >> 5);
    if (e >= E) return;
    int lane = threadIdx.x & 31;
    int s = src[e], d = dst[e];
    const u32b* hrow = (const u32b*)(hB + (size_t)s * 128);
    float* arow = agg + (size_t)d * 128;
    u32b u0 = hrow[lane];
    u32b u1 = hrow[lane + 32];
    atomicAdd(&arow[2 * lane],            bflo(u0));
    atomicAdd(&arow[2 * lane + 1],        bfhi(u0));
    atomicAdd(&arow[2 * (lane + 32)],     bflo(u1));
    atomicAdd(&arow[2 * (lane + 32) + 1], bfhi(u1));
}

// ---------------------------------------------------------- fused dual GEMM
// out[m][n] = sum_k aggB[m][k]*WlT[n][k] + bias[n] + sum_k hB[m][k]*WrT[n][k]
// one wave = one 16x16 output tile, K = 128 (4 wmma k-steps of 32, x2 GEMMs)
__global__ __launch_bounds__(256) void sage_gemm(
    const u16b* __restrict__ aggB, const u16b* __restrict__ hB,
    const u16b* __restrict__ WlT,  const u16b* __restrict__ WrT,
    const float* __restrict__ bias, float* __restrict__ out,
    int Fout, int numTiles, int Nrows) {
    __shared__ u16b sA[16 * 128];
    __shared__ u16b sH[16 * 128];
    const int t = threadIdx.x;
    const int row0 = blockIdx.x * 16;

    // stage both 16x128 bf16 A-tiles (full rows are contiguous): 16B per thread
    const uint4* ga = (const uint4*)(aggB + (size_t)row0 * 128);
    const uint4* gh = (const uint4*)(hB   + (size_t)row0 * 128);
    ((uint4*)sA)[t] = ga[t];
    ((uint4*)sH)[t] = gh[t];
    __syncthreads();

    const int wave = t >> 5, lane = t & 31;
    if (wave >= numTiles) return;           // wave-uniform; EXEC all-1 for WMMA
    const int lo = lane & 15, hi = lane >> 4;
    const int n0 = wave * 16;
    const int n = n0 + lo;

    float bn = (n < Fout) ? bias[n] : 0.0f;
    v8f c;
#pragma unroll
    for (int i = 0; i < 8; i++) c[i] = bn;

    const u16b* wl = WlT + (size_t)n * 128;
    const u16b* wr = WrT + (size_t)n * 128;

#pragma unroll
    for (int kk = 0; kk < 128; kk += 32) {
        Frag a, h, bL, bR;
#pragma unroll
        for (int v = 0; v < 8; v++) {
            // A 16x32 bf16 layout: M=lo, K=(v<4?0:16)+hi*8+2*(v&3)  (+1 in hi half)
            int kb = ((v < 4) ? 0 : 16) + hi * 8 + (v & 3) * 2;
            a.u[v] = *(const u32b*)&sA[lo * 128 + kk + kb];
            h.u[v] = *(const u32b*)&sH[lo * 128 + kk + kb];
            // B 32x16 bf16 layout: N=lo, K=(v<4?0:16)+hi*8+2*(v&3)  (+1 in hi half)
            int kw = kk + kb;
            bL.u[v] = *(const u32b*)&wl[kw];
            bR.u[v] = *(const u32b*)&wr[kw];
        }
        c = __builtin_amdgcn_wmma_f32_16x16x32_bf16(false, a.v, false, bL.v, (short)0, c, false, false);
        c = __builtin_amdgcn_wmma_f32_16x16x32_bf16(false, h.v, false, bR.v, (short)0, c, false, false);
    }

#pragma unroll
    for (int r = 0; r < 8; r++) {
        int m = row0 + hi * 8 + r;          // C/D layout: M = r + 8*hi, N = lo
        if (n < Fout && m < Nrows) out[(size_t)m * Fout + n] = c[r];
    }
}

// -------------------------------------------------------------- batch norm
__global__ __launch_bounds__(256) void bn_stat(
    const float* __restrict__ h, float* __restrict__ sums, float* __restrict__ sumsq,
    int Nrows, int rowsPer) {
    int col = threadIdx.x & 127;
    int half = threadIdx.x >> 7;
    int r0 = blockIdx.x * rowsPer;
    int r1 = r0 + rowsPer; if (r1 > Nrows) r1 = Nrows;
    float s = 0.0f, s2 = 0.0f;
    for (int r = r0 + half; r < r1; r += 2) {
        float v = h[(size_t)r * 128 + col];
        s += v; s2 += v * v;
    }
    __shared__ float ls[256], ls2[256];
    ls[threadIdx.x] = s; ls2[threadIdx.x] = s2;
    __syncthreads();
    if (half == 0) {
        s  += ls[threadIdx.x + 128];
        s2 += ls2[threadIdx.x + 128];
        atomicAdd(&sums[col], s);
        atomicAdd(&sumsq[col], s2);
    }
}

// apply BN (training stats) + ReLU, emit next layer's bf16 activations
__global__ void bn_apply(const float* __restrict__ h,
                         const float* __restrict__ sums, const float* __restrict__ sumsq,
                         const float* __restrict__ g, const float* __restrict__ be,
                         u16b* __restrict__ outB, long long total, float invN) {
    long long i = (long long)blockIdx.x * blockDim.x + threadIdx.x;
    long long stride = (long long)gridDim.x * blockDim.x;
    for (; i < total; i += stride) {
        int col = (int)(i & 127);
        float mean = sums[col] * invN;
        float var  = sumsq[col] * invN - mean * mean;
        float sc   = g[col] * rsqrtf(var + 1e-5f);
        float v = (h[i] - mean) * sc + be[col];
        outB[i] = f2bf(fmaxf(v, 0.0f));
    }
}

// ------------------------------------------------------------- log-softmax
// one wave32 per row of Fout (=40) columns
__global__ __launch_bounds__(256) void log_softmax_rows(
    const float* __restrict__ h, float* __restrict__ out, int Nrows, int Fout) {
    int wave = threadIdx.x >> 5, lane = threadIdx.x & 31;
    int row = blockIdx.x * 8 + wave;
    if (row >= Nrows) return;
    const float* r = h + (size_t)row * Fout;
    float v0 = (lane < Fout) ? r[lane] : -INFINITY;
    float v1 = (lane + 32 < Fout) ? r[lane + 32] : -INFINITY;
    float m = fmaxf(v0, v1);
    for (int o = 16; o; o >>= 1) m = fmaxf(m, __shfl_xor(m, o, 32));
    float s = ((lane < Fout) ? __expf(v0 - m) : 0.0f) +
              ((lane + 32 < Fout) ? __expf(v1 - m) : 0.0f);
    for (int o = 16; o; o >>= 1) s += __shfl_xor(s, o, 32);
    float lse = m + __logf(s);
    float* o0 = out + (size_t)row * Fout;
    if (lane < Fout) o0[lane] = v0 - lse;
    if (lane + 32 < Fout) o0[lane + 32] = v1 - lse;
}

// ---------------------------------------------------------------- launcher
extern "C" void kernel_launch(void* const* d_in, const int* in_sizes, int n_in,
                              void* d_out, int out_size, void* d_ws, size_t ws_size,
                              hipStream_t stream) {
    const float* x    = (const float*)d_in[0];
    const int*   esrc = (const int*)d_in[1];
    const int*   edst = (const int*)d_in[2];
    const float* Wl[3] = { (const float*)d_in[3],  (const float*)d_in[8],  (const float*)d_in[13] };
    const float* bl[3] = { (const float*)d_in[4],  (const float*)d_in[9],  (const float*)d_in[14] };
    const float* Wr[3] = { (const float*)d_in[5],  (const float*)d_in[10], (const float*)d_in[15] };
    const float* g[2]  = { (const float*)d_in[6],  (const float*)d_in[11] };
    const float* be[2] = { (const float*)d_in[7],  (const float*)d_in[12] };

    const int N    = in_sizes[0] / 128;          // 100000
    const int E    = in_sizes[1];                // 1600000
    const int OUTF = in_sizes[13] / 128;         // 40
    const int OUTP = (OUTF + 15) & ~15;          // 48
    const int Npad = (N + 15) & ~15;

    // carve workspace
    char* p = (char*)d_ws;
    auto carve = [&](size_t bytes) -> void* {
        void* r = (void*)p;
        p += (bytes + 255) & ~(size_t)255;
        return r;
    };
    float* deg    = (float*)carve((size_t)N * 4);              // then inv_deg in place
    float* aggF   = (float*)carve((size_t)Npad * 128 * 4);     // fp32 agg accum, reused as GEMM out
    u16b*  aggB   = (u16b*) carve((size_t)Npad * 128 * 2);     // bf16 scaled agg
    u16b*  hB     = (u16b*) carve((size_t)Npad * 128 * 2);     // bf16 activations
    u16b*  WlT[3], *WrT[3];
    for (int l = 0; l < 3; l++) {
        int fp = (l < 2) ? 128 : OUTP;
        WlT[l] = (u16b*)carve((size_t)fp * 128 * 2);
        WrT[l] = (u16b*)carve((size_t)fp * 128 * 2);
    }
    float* sums  = (float*)carve(128 * 4);
    float* sumsq = (float*)carve(128 * 4);

    const long long totalF = (long long)N * 128;

    // degree -> inv_deg
    zero_f32<<<1024, 256, 0, stream>>>(deg, N);
    deg_count<<<(E + 255) / 256, 256, 0, stream>>>(edst, deg, E);
    deg_inv<<<(N + 255) / 256, 256, 0, stream>>>(deg, N);

    // weights: fp32 [K][Fout] -> bf16 [FoutPad][K]
    for (int l = 0; l < 3; l++) {
        int fo = (l < 2) ? 128 : OUTF;
        int fp = (l < 2) ? 128 : OUTP;
        int tot = 128 * fp;
        wprep<<<(tot + 255) / 256, 256, 0, stream>>>(Wl[l], WlT[l], 128, fo, fp);
        wprep<<<(tot + 255) / 256, 256, 0, stream>>>(Wr[l], WrT[l], 128, fo, fp);
    }

    // layer-0 input activations in bf16
    cvt_bf16<<<4096, 256, 0, stream>>>(x, hB, totalF);

    const int gemmGrid = (N + 15) / 16;
    const int rowsPer  = (N + 255) / 256;

    for (int l = 0; l < 3; l++) {
        int fo = (l < 2) ? 128 : OUTF;
        int nt = (l < 2) ? 8 : OUTP / 16;

        // mean-aggregate neighbors
        zero_f32<<<4096, 256, 0, stream>>>(aggF, totalF);
        scatter_add<<<(E + 7) / 8, 256, 0, stream>>>(esrc, edst, hB, aggF, E);
        scale_cvt<<<4096, 256, 0, stream>>>(aggF, deg, aggB, totalF);

        // fused dual GEMM + bias (WMMA bf16, fp32 accum); out -> aggF (reused)
        sage_gemm<<<gemmGrid, 256, 0, stream>>>(aggB, hB, WlT[l], WrT[l],
                                                bl[l], aggF, fo, nt, N);

        if (l < 2) {
            zero_f32<<<1, 128, 0, stream>>>(sums, 128);
            zero_f32<<<1, 128, 0, stream>>>(sumsq, 128);
            bn_stat<<<256, 256, 0, stream>>>(aggF, sums, sumsq, N, rowsPer);
            bn_apply<<<4096, 256, 0, stream>>>(aggF, sums, sumsq, g[l], be[l],
                                               hB, totalF, 1.0f / (float)N);
        } else {
            log_softmax_rows<<<(N + 7) / 8, 256, 0, stream>>>(aggF, (float*)d_out, N, OUTF);
        }
    }
}
